// PointNet2MSG_cls_feature_72877005078944
// MI455X (gfx1250) — matrix operand order repaired and checked by reading, not compiled, not run on hardware
//
#include <hip/hip_runtime.h>

// PointNet++ MSG feature extractor for gfx1250 (MI455X).
// Activations live in WMMA-B-fragment tiled layout so every GEMM operand is a
// contiguous 32B-per-lane vector load (global_load_b128 pairs) and the D tile
// stores straight into next layer's B layout as one v8h (b128) per lane.
// v_wmma_f32_16x16x32_f16, f16 storage, f32 accumulation. Memory-bound.

typedef __attribute__((ext_vector_type(16))) _Float16 v16h;
typedef __attribute__((ext_vector_type(8)))  _Float16 v8h;
typedef __attribute__((ext_vector_type(8)))  float    v8f;

// Tiled half-index for activation element (channel c, column n).
// Layout: ((kb*ntiles + nt)*32 + hi*16 + laneN)*16 + e
//   kb = c/32, hi = (c%32)/16, e = c%16   (K mapping of the B fragment)
//   nt = n/16, laneN = n%16               (N mapping of the B fragment)
__device__ __forceinline__ size_t taddr(int c, long long n, long long ntiles) {
  int kb = c >> 5, rem = c & 31;
  long long nt = n >> 4;
  int laneN = (int)(n & 15);
  return ((((size_t)kb * ntiles + nt) * 32) + ((rem >> 4) * 16 + laneN)) * 16 + (rem & 15);
}

// ---------------------------------------------------------------- FPS -------
__global__ __launch_bounds__(256) void k_fps(const float* __restrict__ pc,
                                             int* __restrict__ fidx, int N, int S) {
  __shared__ float dmin[8192];
  __shared__ float rv[256];
  __shared__ int   ri[256];
  int b = blockIdx.x;
  int t = threadIdx.x;
  const float* p = pc + (size_t)b * N * 3;
  for (int i = t; i < N; i += 256) dmin[i] = 1e10f;
  __syncthreads();
  int far = 0;
  for (int s = 0; s < S; ++s) {
    if (t == 0) fidx[b * S + s] = far;
    float cx = p[far*3+0], cy = p[far*3+1], cz = p[far*3+2];
    float best = -1.0f; int bi = N;
    for (int i = t; i < N; i += 256) {
      float dx = p[i*3+0]-cx, dy = p[i*3+1]-cy, dz = p[i*3+2]-cz;
      float d = dx*dx + dy*dy + dz*dz;
      float dm = fminf(dmin[i], d);
      dmin[i] = dm;
      if (dm > best) { best = dm; bi = i; }   // strict > keeps smallest i per thread
    }
    rv[t] = best; ri[t] = bi;
    __syncthreads();
    for (int off = 128; off > 0; off >>= 1) {
      if (t < off) {
        float ov = rv[t+off]; int oi = ri[t+off];
        if (ov > rv[t] || (ov == rv[t] && oi < ri[t])) { rv[t] = ov; ri[t] = oi; }
      }
      __syncthreads();
    }
    far = ri[0];
    __syncthreads();
  }
}

__global__ void k_gather(const float* __restrict__ pc, const int* __restrict__ idx,
                         float* __restrict__ out, int N, int S, int total) {
  int i = blockIdx.x * blockDim.x + threadIdx.x;
  if (i >= total) return;
  int b = i / S;
  int gi = idx[i];
  const float* src = pc + ((size_t)b * N + gi) * 3;
  out[i*3+0] = src[0]; out[i*3+1] = src[1]; out[i*3+2] = src[2];
}

// ---------------------------------------------------------- ball query ------
__global__ void k_ball(const float* __restrict__ cen, const float* __restrict__ pcb,
                       int* __restrict__ gidx, int N, int S, int K, float r2, int total) {
  int i = blockIdx.x * blockDim.x + threadIdx.x;
  if (i >= total) return;
  int b = i / S;
  const float* p = pcb + (size_t)b * N * 3;
  float cx = cen[i*3+0], cy = cen[i*3+1], cz = cen[i*3+2];
  int* out = gidx + (size_t)i * K;
  int cnt = 0, first = 0;
  for (int j = 0; j < N && cnt < K; ++j) {
    float dx = p[j*3+0]-cx, dy = p[j*3+1]-cy, dz = p[j*3+2]-cz;
    if (dx*dx + dy*dy + dz*dz < r2) {
      if (cnt == 0) first = j;
      out[cnt++] = j;
    }
  }
  for (; cnt < K; ++cnt) out[cnt] = first;
}

// ------------------------------------------------------------- grouping -----
__global__ void k_group1(const float* __restrict__ pc, const float* __restrict__ cen,
                         const int* __restrict__ gidx, _Float16* __restrict__ X,
                         int N, int S, int K, int Cpad, long long Nn, long long ntiles) {
  long long n = (long long)blockIdx.x * blockDim.x + threadIdx.x;
  if (n >= Nn) return;
  long long bs = n / K;
  int b = (int)(bs / S);
  int gi = gidx[n];
  const float* p = pc + ((size_t)b * N + gi) * 3;
  const float* c = cen + (size_t)bs * 3;
  X[taddr(0, n, ntiles)] = (_Float16)(p[0] - c[0]);
  X[taddr(1, n, ntiles)] = (_Float16)(p[1] - c[1]);
  X[taddr(2, n, ntiles)] = (_Float16)(p[2] - c[2]);
  for (int r = 3; r < Cpad; ++r) X[taddr(r, n, ntiles)] = (_Float16)0.f;
}

__global__ void k_group2(const float* __restrict__ pc, const float* __restrict__ cen,
                         const int* __restrict__ gidx, const float* __restrict__ feat,
                         _Float16* __restrict__ X, int N, int S, int K,
                         int Cf, int Sprev, int Cpad, long long Nn, long long ntiles) {
  long long n = (long long)blockIdx.x * blockDim.x + threadIdx.x;
  if (n >= Nn) return;
  long long bs = n / K;
  int b = (int)(bs / S);
  int gi = gidx[n];
  const float* p = pc + ((size_t)b * N + gi) * 3;
  const float* c = cen + (size_t)bs * 3;
  X[taddr(0, n, ntiles)] = (_Float16)(p[0] - c[0]);
  X[taddr(1, n, ntiles)] = (_Float16)(p[1] - c[1]);
  X[taddr(2, n, ntiles)] = (_Float16)(p[2] - c[2]);
  const float* f = feat + (size_t)b * Cf * Sprev + gi;
  for (int cf = 0; cf < Cf; ++cf)
    X[taddr(3 + cf, n, ntiles)] = (_Float16)f[(size_t)cf * Sprev];
  for (int r = 3 + Cf; r < Cpad; ++r) X[taddr(r, n, ntiles)] = (_Float16)0.f;
}

// ------------------------------------------------------------ weights -------
// Swizzle f32 weights into the WMMA A-fragment tiled layout (zero-padded K):
// element e of layout-lane (hi*16+laneM) holds K = hi*16 + e.
__global__ void k_padw(const float* __restrict__ W, _Float16* __restrict__ Wt,
                       int Cout, int Cin, int Cinp) {
  int i = blockIdx.x * blockDim.x + threadIdx.x;
  if (i >= Cout * Cinp) return;
  int o = i / Cinp, k = i % Cinp;
  float v = (k < Cin) ? W[o * Cin + k] : 0.f;
  int mt = o >> 4, laneM = o & 15;
  int kb = k >> 5, kk = k & 31;
  int hi = kk >> 4, e = kk & 15;
  Wt[(((size_t)mt * (Cinp >> 5) + kb) * 32 + hi * 16 + laneM) * 16 + e] = (_Float16)v;
}

// --------------------------------------------------------------- GEMM -------
// Y[Cout,N] = W[Cout,Cin] * X[Cin,N]. 8 waves/block, each wave owns one n-tile
// and two m-tiles (MT=2: B fragment reused -> X traffic halved). Operands are
// contiguous v16h per lane (2x global_load_b128); D stores as one v8h (b128)
// directly in next layer's B-fragment layout.
__global__ __launch_bounds__(256) void k_gemm(const _Float16* __restrict__ Wt,
                                              const _Float16* __restrict__ Xt,
                                              _Float16* __restrict__ Yt,
                                              int KB, long long ntiles) {
  int lane = threadIdx.x & 31;
  int wave = threadIdx.x >> 5;
  long long nt = (long long)blockIdx.x * 8 + wave;
  int mt0 = blockIdx.y * 2;

  v8f acc0 = {}, acc1 = {};
  const v16h* xb = (const v16h*)Xt + (nt * 32 + lane);
  const v16h* wb = (const v16h*)Wt + lane;
  size_t xstep = (size_t)ntiles * 32;             // v16h units per kb
  size_t w0 = (size_t)(mt0 + 0) * KB * 32;
  size_t w1 = (size_t)(mt0 + 1) * KB * 32;

  for (int kb = 0; kb < KB; ++kb) {
    v16h bf = xb[(size_t)kb * xstep];
    if (kb + 1 < KB)
      __builtin_prefetch((const void*)(xb + (size_t)(kb + 1) * xstep), 0, 0);
    v16h a0 = wb[w0 + (size_t)kb * 32];
    acc0 = __builtin_amdgcn_wmma_f32_16x16x32_f16(
        false, a0, false, bf, (short)0, acc0, false, false);
    v16h a1 = wb[w1 + (size_t)kb * 32];
    acc1 = __builtin_amdgcn_wmma_f32_16x16x32_f16(
        false, a1, false, bf, (short)0, acc1, false, false);
  }

  int laneN = lane & 15, hi = lane >> 4;
#pragma unroll
  for (int m = 0; m < 2; ++m) {
    int mt = mt0 + m;
    v8f acc = m ? acc1 : acc0;
    v8h out;
#pragma unroll
    for (int r = 0; r < 8; ++r) out[r] = (_Float16)acc[r];
    // c = mt*16 + hi*8 + r  ->  kb' = mt>>1, hi' = mt&1, e' = hi*8+r (contig.)
    _Float16* dst = Yt + ((((size_t)(mt >> 1) * ntiles + nt) * 32)
                          + (mt & 1) * 16 + laneN) * 16 + hi * 8;
    *(v8h*)dst = out;
  }
}

// ------------------------------------------------------------ batchnorm -----
__global__ __launch_bounds__(256) void k_bnstats(const _Float16* __restrict__ Y,
                                                 float* __restrict__ mean,
                                                 float* __restrict__ rstd,
                                                 long long Nn, long long ntiles) {
  __shared__ float r1[256], r2[256];
  int c = blockIdx.x;
  int kb = c >> 5, rem = c & 31;
  size_t cofs = (size_t)((rem >> 4) * 16) * 16 + (rem & 15);
  const _Float16* y = Y + (size_t)kb * ntiles * 512;
  float s1 = 0.f, s2 = 0.f;
  for (long long i = threadIdx.x; i < Nn; i += 256) {
    long long nt = i >> 4;
    int laneN = (int)(i & 15);
    float v = (float)y[nt * 512 + cofs + (size_t)laneN * 16];
    s1 += v; s2 += v * v;
  }
  r1[threadIdx.x] = s1; r2[threadIdx.x] = s2;
  __syncthreads();
  for (int off = 128; off > 0; off >>= 1) {
    if (threadIdx.x < off) {
      r1[threadIdx.x] += r1[threadIdx.x + off];
      r2[threadIdx.x] += r2[threadIdx.x + off];
    }
    __syncthreads();
  }
  if (threadIdx.x == 0) {
    float m = r1[0] / (float)Nn;
    float v = r2[0] / (float)Nn - m * m;
    mean[c] = m;
    rstd[c] = rsqrtf(v + 1e-5f);
  }
}

// In-place normalize+ReLU over the tiled layout; channel recovered by shifts
// (ntiles is always a power of two here).
__global__ void k_bnrelu(_Float16* __restrict__ Y, const float* __restrict__ mean,
                         const float* __restrict__ rstd, const float* __restrict__ g,
                         const float* __restrict__ bb, int ntshift, long long total) {
  long long i = (long long)blockIdx.x * blockDim.x + threadIdx.x;
  if (i >= total) return;
  int e  = (int)(i & 15);
  int hi = (int)((i >> 8) & 1);
  long long blk = i >> 9;
  int kb = (int)(blk >> ntshift);
  int c = kb * 32 + hi * 16 + e;
  float v = (float)Y[i];
  v = (v - mean[c]) * rstd[c] * g[c] + bb[c];
  Y[i] = (_Float16)fmaxf(v, 0.f);
}

// ------------------------------------------------------------- pooling ------
__global__ void k_maxpool(const _Float16* __restrict__ Y, float* __restrict__ F,
                          int S, int K, int C, int coff, int Ct, long long ntiles,
                          long long total) {
  long long i = (long long)blockIdx.x * blockDim.x + threadIdx.x;
  if (i >= total) return;
  int c = (int)(i % C);
  long long bs = i / C;
  int b = (int)(bs / S), s = (int)(bs % S);
  float m = -3.4e38f;
  for (int j = 0; j < K; ++j)
    m = fmaxf(m, (float)Y[taddr(c, bs * K + j, ntiles)]);
  F[((size_t)b * Ct + coff + c) * S + s] = m;
}

// ---------------------------------------------------------------- SA --------
__global__ void k_sa_build(const float* __restrict__ xyz2, const float* __restrict__ f2,
                           _Float16* __restrict__ X, long long total) {
  long long i = (long long)blockIdx.x * blockDim.x + threadIdx.x;
  if (i >= total) return;            // total = 672 * 2048; n = b*128+s
  int c = (int)(i / 2048);
  int n = (int)(i % 2048);
  int b = n >> 7, s = n & 127;
  float v = 0.f;
  if (c < 3)        v = xyz2[((size_t)b * 128 + s) * 3 + c];
  else if (c < 643) v = f2[((size_t)b * 640 + (c - 3)) * 128 + s];
  X[taddr(c, n, 128)] = (_Float16)v;
}

__global__ void k_final_max(const _Float16* __restrict__ Y, float* __restrict__ out,
                            int total) {
  int i = blockIdx.x * blockDim.x + threadIdx.x;
  if (i >= total) return;            // total = 16*1024, out[b*1024+c]
  int b = i / 1024, c = i % 1024;
  float m = -3.4e38f;
  for (int s = 0; s < 128; ++s)
    m = fmaxf(m, (float)Y[taddr(c, (long long)b * 128 + s, 128)]);
  out[i] = m;
}

// ------------------------------------------------------------- host side ----
static const size_t OFF_FPS1 = 0;                            // 16*512*4
static const size_t OFF_XYZ1 = OFF_FPS1 + 32768;             // 16*512*3*4
static const size_t OFF_FPS2 = OFF_XYZ1 + 98304;             // 16*128*4
static const size_t OFF_XYZ2 = OFF_FPS2 + 8192;              // 16*128*3*4
static const size_t OFF_F1   = OFF_XYZ2 + 24576;             // 16*320*512*4
static const size_t OFF_F2   = OFF_F1 + 10485760;            // 16*640*128*4
static const size_t OFF_GIDX = OFF_F2 + 5242880;             // 16*512*128*4
static const size_t OFF_WPAD = OFF_GIDX + 4194304;           // 1024*672*2
static const size_t OFF_MEAN = OFF_WPAD + 1376256;           // 1024*4
static const size_t OFF_RSTD = OFF_MEAN + 4096;              // 1024*4
static const size_t OFF_BUF0 = OFF_RSTD + 4096;              // 128*1M f16
static const size_t OFF_BUF1 = OFF_BUF0 + 268435456;         // 128*1M f16
// total workspace need: ~558 MB

struct LP { const float *W, *g, *b; };

static void run_layer(const LP& lp, int Cout, int Cin, int Cinp, long long Nn,
                      const _Float16* X, _Float16* Y, _Float16* Wp,
                      float* mean, float* rstd, hipStream_t stream) {
  long long ntiles = Nn >> 4;
  int ntshift = __builtin_ctzll((unsigned long long)ntiles);
  int wtot = Cout * Cinp;
  k_padw<<<(wtot + 255) / 256, 256, 0, stream>>>(lp.W, Wp, Cout, Cin, Cinp);
  k_gemm<<<dim3((unsigned)(ntiles / 8), (unsigned)(Cout / 32)), 256, 0, stream>>>(
      Wp, X, Y, Cinp >> 5, ntiles);
  k_bnstats<<<Cout, 256, 0, stream>>>(Y, mean, rstd, Nn, ntiles);
  long long tot = (long long)Cout * Nn;
  k_bnrelu<<<(unsigned)((tot + 255) / 256), 256, 0, stream>>>(
      Y, mean, rstd, lp.g, lp.b, ntshift, tot);
}

extern "C" void kernel_launch(void* const* d_in, const int* in_sizes, int n_in,
                              void* d_out, int out_size, void* d_ws, size_t ws_size,
                              hipStream_t stream) {
  (void)in_sizes; (void)n_in; (void)out_size; (void)ws_size;
  const float* P[64];
  for (int i = 0; i < 64; ++i) P[i] = (const float*)d_in[i];
  const float* pc = P[0];

  LP msg1[3][3], msg2[3][3], sa[3];
  int q = 1;
  for (int br = 0; br < 3; ++br)
    for (int l = 0; l < 3; ++l) { msg1[br][l] = {P[q], P[q+1], P[q+2]}; q += 3; }
  for (int br = 0; br < 3; ++br)
    for (int l = 0; l < 3; ++l) { msg2[br][l] = {P[q], P[q+1], P[q+2]}; q += 3; }
  for (int l = 0; l < 3; ++l) { sa[l] = {P[q], P[q+1], P[q+2]}; q += 3; }

  static const int d1[3][4] = {{3,32,32,64},{3,64,64,128},{3,64,96,128}};
  static const int d2[3][4] = {{323,64,64,128},{323,128,128,256},{323,128,128,256}};

  char* ws = (char*)d_ws;
  int*      fps1 = (int*)     (ws + OFF_FPS1);
  float*    xyz1 = (float*)   (ws + OFF_XYZ1);
  int*      fps2 = (int*)     (ws + OFF_FPS2);
  float*    xyz2 = (float*)   (ws + OFF_XYZ2);
  float*    f1   = (float*)   (ws + OFF_F1);
  float*    f2   = (float*)   (ws + OFF_F2);
  int*      gidx = (int*)     (ws + OFF_GIDX);
  _Float16* Wp   = (_Float16*)(ws + OFF_WPAD);
  float*    mean = (float*)   (ws + OFF_MEAN);
  float*    rstd = (float*)   (ws + OFF_RSTD);
  _Float16* buf0 = (_Float16*)(ws + OFF_BUF0);
  _Float16* buf1 = (_Float16*)(ws + OFF_BUF1);

  const int B = 16;

  // ---- MSG layer 1: N=8192, S=512 ----
  k_fps<<<B, 256, 0, stream>>>(pc, fps1, 8192, 512);
  k_gather<<<(B*512 + 255) / 256, 256, 0, stream>>>(pc, fps1, xyz1, 8192, 512, B*512);
  {
    const float radii[3] = {0.1f, 0.2f, 0.4f};
    const int   ks[3]    = {16, 32, 128};
    int coff = 0;
    for (int br = 0; br < 3; ++br) {
      int K = ks[br];
      long long Nn = (long long)B * 512 * K;
      long long ntiles = Nn >> 4;
      k_ball<<<(B*512 + 255) / 256, 256, 0, stream>>>(
          xyz1, pc, gidx, 8192, 512, K, radii[br]*radii[br], B*512);
      k_group1<<<(unsigned)((Nn + 255) / 256), 256, 0, stream>>>(
          pc, xyz1, gidx, buf0, 8192, 512, K, 32, Nn, ntiles);
      int Cin = 3, Cinp = 32;
      _Float16 *X = buf0, *Y = buf1;
      for (int l = 0; l < 3; ++l) {
        int Cout = d1[br][l+1];
        run_layer(msg1[br][l], Cout, Cin, Cinp, Nn, X, Y, Wp, mean, rstd, stream);
        _Float16* t = X; X = Y; Y = t;
        Cin = Cout; Cinp = Cout;
      }
      // after 3 layers the (normalized) output sits in buf1
      int Clast = d1[br][3];
      long long ptot = (long long)B * 512 * Clast;
      k_maxpool<<<(unsigned)((ptot + 255) / 256), 256, 0, stream>>>(
          buf1, f1, 512, K, Clast, coff, 320, ntiles, ptot);
      coff += Clast;
    }
  }

  // ---- MSG layer 2: N=512, S=128, feat = f1 (320ch) ----
  k_fps<<<B, 256, 0, stream>>>(xyz1, fps2, 512, 128);
  k_gather<<<(B*128 + 255) / 256, 256, 0, stream>>>(xyz1, fps2, xyz2, 512, 128, B*128);
  {
    const float radii[3] = {0.2f, 0.4f, 0.8f};
    const int   ks[3]    = {32, 64, 128};
    int coff = 0;
    for (int br = 0; br < 3; ++br) {
      int K = ks[br];
      long long Nn = (long long)B * 128 * K;
      long long ntiles = Nn >> 4;
      k_ball<<<(B*128 + 255) / 256, 256, 0, stream>>>(
          xyz2, xyz1, gidx, 512, 128, K, radii[br]*radii[br], B*128);
      k_group2<<<(unsigned)((Nn + 255) / 256), 256, 0, stream>>>(
          xyz1, xyz2, gidx, f1, buf0, 512, 128, K, 320, 512, 352, Nn, ntiles);
      int Cin = 323, Cinp = 352;
      _Float16 *X = buf0, *Y = buf1;
      for (int l = 0; l < 3; ++l) {
        int Cout = d2[br][l+1];
        run_layer(msg2[br][l], Cout, Cin, Cinp, Nn, X, Y, Wp, mean, rstd, stream);
        _Float16* t = X; X = Y; Y = t;
        Cin = Cout; Cinp = Cout;
      }
      int Clast = d2[br][3];
      long long ptot = (long long)B * 128 * Clast;
      k_maxpool<<<(unsigned)((ptot + 255) / 256), 256, 0, stream>>>(
          buf1, f2, 128, K, Clast, coff, 640, ntiles, ptot);
      coff += Clast;
    }
  }

  // ---- SA head: X = concat(xyz2^T, f2) -> 643 ch (pad 672), n = 16*128 ----
  {
    long long Nn = 2048;
    long long btot = 672LL * 2048;
    k_sa_build<<<(unsigned)((btot + 255) / 256), 256, 0, stream>>>(xyz2, f2, buf0, btot);
    static const int sdim[4] = {643, 256, 512, 1024};
    int Cin = 643, Cinp = 672;
    _Float16 *X = buf0, *Y = buf1;
    for (int l = 0; l < 3; ++l) {
      int Cout = sdim[l+1];
      run_layer(sa[l], Cout, Cin, Cinp, Nn, X, Y, Wp, mean, rstd, stream);
      _Float16* t = X; X = Y; Y = t;
      Cin = Cout; Cinp = Cout;
    }
    // final activations in buf1: [1024][2048]; max over s=128 per batch
    k_final_max<<<(16*1024 + 255) / 256, 256, 0, stream>>>(buf1, (float*)d_out, 16*1024);
  }
}
